// CRF_69707319214308
// MI455X (gfx1250) — matrix operand order repaired and checked by reading, not compiled, not run on hardware
//
#include <hip/hip_runtime.h>

typedef __attribute__((ext_vector_type(16))) _Float16 v16h;
typedef __attribute__((ext_vector_type(8)))  _Float16 v8h;
typedef __attribute__((ext_vector_type(8)))  float    v8f;
typedef __attribute__((ext_vector_type(4)))  float    v4f;

#define S_LEN 512
#define BATCH 1024
#define NTAG  64
#define LOG2E 1.4426950408889634f
#define LN2   0.6931471805599453f

__device__ __forceinline__ float log2_fast(float x) { return __builtin_amdgcn_logf(x); }   // v_log_f32
__device__ __forceinline__ float exp2_fast(float x) { return __builtin_amdgcn_exp2f(x); }  // v_exp_f32

// ---------------------------------------------------------------------------
// LDS relayout between the WMMA D layout (q^T) and the next step's B layout.
// Lane mapping (wave32): col = lane&15 -> batch (D and B columns coincide),
// h = lane>>4. D frag mc, slot r : tag = 16*mc + r + 8*h.
// B frag kc, elem e : tag = 32*kc + 16*h + e. A row-major f16 image
// p[batch][tag] in LDS converts one to the other with contiguous b128 ops.
// ---------------------------------------------------------------------------
__device__ __forceinline__ void store_u(_Float16* pbuf, int col, int h,
                                        const float u[4][8]) {
#pragma unroll
  for (int mc = 0; mc < 4; ++mc) {
    v8h hv;
#pragma unroll
    for (int r = 0; r < 8; ++r) hv[r] = (_Float16)u[mc][r];
    *(v8h*)(pbuf + col * NTAG + 16 * mc + 8 * h) = hv;   // ds_store_b128
  }
}

__device__ __forceinline__ void load_b(const _Float16* pbuf, int col, int h,
                                       v16h bf[2]) {
#pragma unroll
  for (int kc = 0; kc < 2; ++kc) {
    const v8h lo = *(const v8h*)(pbuf + col * NTAG + 32 * kc + 16 * h);      // ds_load_b128
    const v8h hi = *(const v8h*)(pbuf + col * NTAG + 32 * kc + 16 * h + 8);  // ds_load_b128
#pragma unroll
    for (int i = 0; i < 8; ++i) { bf[kc][i] = lo[i]; bf[kc][i + 8] = hi[i]; }
  }
}

// ---------------------------------------------------------------------------
// CRF log-partition via WMMA, scaled-forward in probability space with
// *deferred* normalization: state u is unnormalized f16, invariant
// alpha = ln(u) + L*ln2. Scale t = log2(sum u) comes from the previous
// iteration, so the per-step critical path is only
//   ds_load -> wmma -> pk_mul -> cvt_pk -> ds_store.
// ---------------------------------------------------------------------------
__global__ __launch_bounds__(32)
void crf_logz_wmma(const float* __restrict__ em,
                   const float* __restrict__ trans,
                   const float* __restrict__ startt,
                   const float* __restrict__ endt,
                   const int*   __restrict__ mask,
                   float*       __restrict__ out) {
  __shared__ alignas(32) _Float16 pbuf[16 * NTAG];

  const int lane = threadIdx.x;
  const int col  = lane & 15;
  const int h    = lane >> 4;
  const int batch = blockIdx.x * 16 + col;

  // ---- Constant A fragments: A = E^T, E[i][j] = exp(trans[i][j]) ----
  // A frag (mc,kc): row M = out-tag 16*mc + col; elem e -> K = in-tag:
  //   K = 32*kc + 16*h + e
  v16h af[4][2];
#pragma unroll
  for (int mc = 0; mc < 4; ++mc)
#pragma unroll
    for (int kc = 0; kc < 2; ++kc)
#pragma unroll
      for (int e = 0; e < 16; ++e) {
        int v  = e >> 1, w = e & 1;
        int kb = (v < 4) ? (2 * v) : (2 * v + 8);
        int i_in  = 32 * kc + 16 * h + e;
        int j_out = 16 * mc + col;
        af[mc][kc][e] = (_Float16)exp2_fast(trans[i_in * NTAG + j_out] * LOG2E);
      }

  float L;      // per-batch log2 scale (replicated across halves)
  float cst;    // sum of current (unnormalized) u, f32
  v16h  bf[2];  // u^T in B-operand layout (f16, unnormalized)

  // ---- Init: u0 = exp(start + em[0]) (fits f16), L = 0 ----
  {
    const float* ep = em + (size_t)batch * NTAG;
    float u[4][8];
    float part = 0.f;
#pragma unroll
    for (int mc = 0; mc < 4; ++mc) {
      int t0 = 16 * mc + 8 * h;
      v4f e0 = __builtin_nontemporal_load((const v4f*)(ep + t0));
      v4f e1 = __builtin_nontemporal_load((const v4f*)(ep + t0 + 4));
      v4f s0 = *(const v4f*)(startt + t0);
      v4f s1 = *(const v4f*)(startt + t0 + 4);
#pragma unroll
      for (int r = 0; r < 4; ++r) {
        u[mc][r]     = exp2_fast((e0[r] + s0[r]) * LOG2E);
        u[mc][r + 4] = exp2_fast((e1[r] + s1[r]) * LOG2E);
      }
#pragma unroll
      for (int r = 0; r < 8; ++r) part += u[mc][r];
    }
    cst = part + __shfl_xor(part, 16, 32);
    L   = 0.f;
    store_u(pbuf, col, h, u);
    load_b(pbuf, col, h, bf);
  }

  // ---- Recurrence ----
  for (int s = 1; s < S_LEN; ++s) {
    const float* ep = em + ((size_t)s * BATCH + batch) * NTAG;
    const int m = mask[s * BATCH + batch];
    v4f e0[4], e1[4];
#pragma unroll
    for (int mc = 0; mc < 4; ++mc) {
      int t0 = 16 * mc + 8 * h;
      e0[mc] = __builtin_nontemporal_load((const v4f*)(ep + t0));      // b128 NT
      e1[mc] = __builtin_nontemporal_load((const v4f*)(ep + t0 + 4));  // b128 NT
    }

    // Scale from PREVIOUS iteration's sum: off the WMMA critical path.
    const float t = log2_fast(cst);

    // wr = exp2(em*log2e - t) : one fma + one v_exp per element
    float wr[4][8];
#pragma unroll
    for (int mc = 0; mc < 4; ++mc)
#pragma unroll
      for (int r = 0; r < 4; ++r) {
        wr[mc][r]     = exp2_fast(fmaf(e0[mc][r], LOG2E, -t));
        wr[mc][r + 4] = exp2_fast(fmaf(e1[mc][r], LOG2E, -t));
      }

    // q^T = E^T x u^T : 8 WMMAs
    v8f dd[4];
#pragma unroll
    for (int mc = 0; mc < 4; ++mc) {
      v8f z = {};
      z = __builtin_amdgcn_wmma_f32_16x16x32_f16(false, af[mc][0], false, bf[0],
                                                 (short)0, z, false, false);
      z = __builtin_amdgcn_wmma_f32_16x16x32_f16(false, af[mc][1], false, bf[1],
                                                 (short)0, z, false, false);
      dd[mc] = z;
    }

    // u' = q * wr ; store immediately (critical path), sum in the shadow
    float u[4][8];
#pragma unroll
    for (int mc = 0; mc < 4; ++mc)
#pragma unroll
      for (int r = 0; r < 8; ++r)
        u[mc][r] = dd[mc][r] * wr[mc][r];
    store_u(pbuf, col, h, u);

    float part = 0.f;
#pragma unroll
    for (int mc = 0; mc < 4; ++mc)
#pragma unroll
      for (int r = 0; r < 8; ++r) part += u[mc][r];
    float cn = part + __shfl_xor(part, 16, 32);

    // masked rows keep old state (u, scale, sum)
    L   = m ? (L + t) : L;
    cst = m ? cn : cst;
    v16h bn[2];
    load_b(pbuf, col, h, bn);
    bf[0] = m ? bn[0] : bf[0];
    bf[1] = m ? bn[1] : bf[1];
  }

  // ---- log Z = ln2 * ( log2( sum_j u_j * exp(end_j) ) + L ) ----
  float part = 0.f;
#pragma unroll
  for (int kc = 0; kc < 2; ++kc) {
    int t0 = 32 * kc + 16 * h;
#pragma unroll
    for (int e = 0; e < 16; ++e)
      part += (float)bf[kc][e] * exp2_fast(endt[t0 + e] * LOG2E);
  }
  float tot  = part + __shfl_xor(part, 16, 32);
  float logz = (log2_fast(tot) + L) * LN2;
  if (lane < 16) atomicAdd(out, logz);
}

// ---------------------------------------------------------------------------
// Gold-path score.
// ---------------------------------------------------------------------------
__global__ void crf_score(const float* __restrict__ em,
                          const float* __restrict__ trans,
                          const float* __restrict__ startt,
                          const float* __restrict__ endt,
                          const int*   __restrict__ tags,
                          const int*   __restrict__ mask,
                          float*       __restrict__ out) {
  int b = blockIdx.x * blockDim.x + threadIdx.x;
  if (b >= BATCH) return;
  int tp = tags[b];
  float sc = startt[tp] + em[(size_t)b * NTAG + tp];
  int cnt = mask[b];
  for (int s = 1; s < S_LEN; ++s) {
    int t = tags[s * BATCH + b];
    int m = mask[s * BATCH + b];
    cnt += m;
    float add = trans[tp * NTAG + t] + em[((size_t)s * BATCH + b) * NTAG + t];
    sc += m ? add : 0.f;
    tp = t;
  }
  if (cnt < 1) cnt = 1;
  int last = tags[(cnt - 1) * BATCH + b];
  sc += endt[last];
  atomicAdd(out, -sc);
}

__global__ void crf_zero(float* out) {
  if (threadIdx.x == 0 && blockIdx.x == 0) *out = 0.f;
}

extern "C" void kernel_launch(void* const* d_in, const int* in_sizes, int n_in,
                              void* d_out, int out_size, void* d_ws, size_t ws_size,
                              hipStream_t stream) {
  const float* em    = (const float*)d_in[0];  // (S,B,T)
  const float* trans = (const float*)d_in[1];  // (T,T)
  const float* st    = (const float*)d_in[2];  // (T,)
  const float* en    = (const float*)d_in[3];  // (T,)
  const int*   tags  = (const int*)d_in[4];    // (S,B)
  const int*   mask  = (const int*)d_in[5];    // (S,B)
  float* out = (float*)d_out;

  crf_zero<<<1, 32, 0, stream>>>(out);
  crf_score<<<(BATCH + 255) / 256, 256, 0, stream>>>(em, trans, st, en, tags, mask, out);
  crf_logz_wmma<<<BATCH / 16, 32, 0, stream>>>(em, trans, st, en, mask, out);
}
